// PerformerSelfAttention_74131135529808
// MI455X (gfx1250) — compile-verified, matrix operand order checked by reading, not run on hardware
//
#include <hip/hip_runtime.h>

// ---------------------------------------------------------------------------
// Problem constants (from reference)
// ---------------------------------------------------------------------------
constexpr int   Bb   = 8;
constexpr int   Cc   = 256;
constexpr int   Nn   = 4096;          // SIZE*SIZE
constexpr int   Hh   = 4;
constexpr int   Dd   = 64;
constexpr int   Mv   = 266;           // valid random features
constexpr int   Mp   = 272;           // padded to multiple of 16
constexpr int   VE   = 80;            // V extended width: 64 v-cols + 1 ones + pad
constexpr long long BNrows = (long long)Bb * Nn;   // 32768
constexpr int   BH   = Bb * Hh;       // 32
constexpr float LN_EPS   = 1e-5f;
constexpr float KERN_EPS = 1e-4f;
constexpr float NORMALIZER = 0.35355339059327373f;   // 64^-0.25
constexpr float RATIO      = 0.06131393394849658f;   // 266^-0.5

typedef __attribute__((ext_vector_type(16))) _Float16 v16h;
typedef __attribute__((ext_vector_type(8)))  _Float16 v8h;
typedef __attribute__((ext_vector_type(2)))  _Float16 h2v;
typedef __attribute__((ext_vector_type(8)))  float    v8f;

__device__ __forceinline__ unsigned ordf(float f) {
  unsigned u = __float_as_uint(f);
  return u ^ ((u >> 31) ? 0xFFFFFFFFu : 0x80000000u);
}
__device__ __forceinline__ float unordf(unsigned o) {
  unsigned u = o ^ ((o >> 31) ? 0x80000000u : 0xFFFFFFFFu);
  return __uint_as_float(u);
}
__device__ __forceinline__ unsigned umaxu(unsigned a, unsigned b) { return a > b ? a : b; }

// wait for all outstanding async-to-LDS transfers from this wave
__device__ __forceinline__ void wait_async_lds() {
#if defined(__has_builtin)
#if __has_builtin(__builtin_amdgcn_s_wait_asynccnt)
  __builtin_amdgcn_s_wait_asynccnt(0);
  return;
#endif
#endif
  asm volatile("s_wait_asynccnt 0x0" ::: "memory");
}

// ---------------------------------------------------------------------------
// Generic WMMA f16 GEMM:  D = epilogue(A[M,K] * B[K,N])
//  - block = 128 threads = 4 wave32, block tile 64x64, K-step 32
//  - A,B staged in LDS; B transposed at LDS store so both fragment loads are
//    contiguous 16B ds reads matching the CDNA5 16-bit A/B VGPR layout
//  - !ABND && !TRANSA: A tile staged with global_load_async_to_lds_b128
//    (ASYNCcnt), unconditional j-indexed issues (no exec-mask juggling)
//  - bounded sides: branchless clamp+select; staging split into a load phase
//    (all vector loads in flight) and a store phase (single loadcnt wait)
//  - TRANSA: A logical [M,K] read from Asrc[K,lda] (used for kp^T * Vext)
//  Epilogues: 0 f16 store *alpha | 1 f32 store (+ordered-u32 atomic row max)
//             2 f32 +bias+resid  | 3 f16 gelu(x+bias) | 4 final: +bias+resid,
//             transposed scatter into [B,C,N] output
// ---------------------------------------------------------------------------
#define TBM 64
#define TBN 64
#define TBK 32
#define LDP 40   // LDS pitch in halves (80B rows: 16B-aligned frag chunks)

template<int EP, bool TRANSA, bool ABND, bool BBND>
__global__ __launch_bounds__(128) void wmma_gemm(
    const _Float16* __restrict__ A, const _Float16* __restrict__ Bmat,
    int M, int Ncols, int K,
    int lda, int ldb, int ldd,
    long long sAb, long long sAh, long long sBz, long long sDz, int Hn,
    float alpha,
    float* __restrict__ Df32, _Float16* __restrict__ Df16,
    const float* __restrict__ bias,
    const float* __restrict__ resid, int ldres,
    unsigned* __restrict__ rowmax, long long sRowz, int Nvalid,
    float* __restrict__ finalOut, int nPerB)
{
  __shared__ __align__(16) _Float16 As[TBM * LDP];
  __shared__ __align__(16) _Float16 Bs[TBN * LDP];

  const int tid  = threadIdx.x;
  const int lane = tid & 31;
  const int wave = tid >> 5;
  const int wm   = wave >> 1;          // wave row (0..1) -> 32 rows
  const int wn   = wave & 1;           // wave col (0..1) -> 32 cols
  const int lh   = lane & 15;          // lane within 16
  const int hs   = lane >> 4;          // half-select

  const int z  = blockIdx.z;
  const long long Aoff = (long long)(z / Hn) * sAb + (long long)(z % Hn) * sAh;
  const long long Boff = (long long)z * sBz;
  const long long Doff = (long long)z * sDz;
  const int m0 = blockIdx.y * TBM;
  const int n0 = blockIdx.x * TBN;
  const _Float16 hzero = (_Float16)0.f;

  v8f acc[2][2];
  for (int i = 0; i < 2; ++i)
    for (int j = 0; j < 2; ++j)
      for (int r = 0; r < 8; ++r) acc[i][j][r] = 0.f;

  for (int kk = 0; kk < K; kk += TBK) {
    // ---------------- A tile -> As[m][k] ----------------
    if (!TRANSA && !ABND) {
      // async DMA: 2 unconditional 16B chunks per thread, ASYNCcnt-tracked
#pragma unroll
      for (int j = 0; j < 2; ++j) {
        int i = tid + j * 128;
        int row = i >> 2, c = (i & 3) << 3;              // 8 halves = 16B
        const _Float16* gp = A + Aoff + (long long)(m0 + row) * lda + kk + c;
        unsigned lb = (unsigned)(size_t)(const void*)&As[row * LDP + c];
        asm volatile("global_load_async_to_lds_b128 %0, %1, off"
                     :: "v"(lb), "v"(gp) : "memory");
      }
    } else if (!TRANSA) {
      h2v av[8];
#pragma unroll
      for (int j = 0; j < 8; ++j) {                      // load phase
        int i = tid + j * 128;
        int row = i >> 4, cp = (i & 15) << 1;
        int gm = m0 + row, gk = kk + cp;
        bool ok = (gm < M) && (gk < K);
        long long o = ok ? ((long long)gm * lda + gk) : 0;
        h2v val = *(const h2v*)(A + Aoff + o);
        val[0] = ok ? val[0] : hzero;
        val[1] = ok ? val[1] : hzero;
        av[j] = val;
      }
#pragma unroll
      for (int j = 0; j < 8; ++j) {                      // store phase
        int i = tid + j * 128;
        int row = i >> 4, cp = (i & 15) << 1;
        *(h2v*)(&As[row * LDP + cp]) = av[j];
      }
    } else {
      h2v av[8];
#pragma unroll
      for (int j = 0; j < 8; ++j) {                      // pairs along m
        int i = tid + j * 128;
        int k = i >> 5, mp = (i & 31) << 1;
        int gm = m0 + mp, gk = kk + k;
        bool ok = (gm + 2 <= M) && (gk < K);             // M always even here
        long long o = ok ? ((long long)gk * lda + gm) : 0;
        h2v val = *(const h2v*)(A + Aoff + o);
        val[0] = ok ? val[0] : hzero;
        val[1] = ok ? val[1] : hzero;
        av[j] = val;
      }
#pragma unroll
      for (int j = 0; j < 8; ++j) {
        int i = tid + j * 128;
        int k = i >> 5, mp = (i & 31) << 1;
        As[mp * LDP + k]       = av[j][0];
        As[(mp + 1) * LDP + k] = av[j][1];
      }
    }

    // ---------------- B tile -> Bs[n][k] (transpose at store) ----------------
    {
      h2v bv[8];
#pragma unroll
      for (int j = 0; j < 8; ++j) {                      // load phase
        int i = tid + j * 128;
        int k = i >> 5, np = (i & 31) << 1;
        if (!BBND) {
          bv[j] = *(const h2v*)(Bmat + Boff + (long long)(kk + k) * ldb + n0 + np);
        } else {
          int gk = kk + k, gn = n0 + np;
          bool ok = (gk < K) && (gn + 2 <= Ncols);       // Ncols always even
          long long o = ok ? ((long long)gk * ldb + gn) : 0;
          h2v val = *(const h2v*)(Bmat + Boff + o);
          val[0] = ok ? val[0] : hzero;
          val[1] = ok ? val[1] : hzero;
          bv[j] = val;
        }
      }
#pragma unroll
      for (int j = 0; j < 8; ++j) {                      // store phase
        int i = tid + j * 128;
        int k = i >> 5, np = (i & 31) << 1;
        Bs[np * LDP + k]       = bv[j][0];
        Bs[(np + 1) * LDP + k] = bv[j][1];
      }
    }
    if (!TRANSA && !ABND) wait_async_lds();
    __syncthreads();

    // prefetch next K tile into cache (global_prefetch_b8)
    if (kk + TBK < K) {
      if (!TRANSA)
        __builtin_prefetch(A + Aoff + (long long)(m0 + (tid & 63)) * lda + kk + TBK, 0, 1);
      else
        __builtin_prefetch(A + Aoff + (long long)(kk + TBK + (tid & 31)) * lda + m0, 0, 1);
      __builtin_prefetch(Bmat + Boff + (long long)(kk + TBK + (tid & 31)) * ldb + n0, 0, 1);
    }

    // ---- fragments: K = (e/8)*16 + hs*8 + (e%8); lane -> M (A) / N (B) ----
    v16h afr[2], bfr[2];
    for (int t = 0; t < 2; ++t) {
      int mrow = wm * 32 + t * 16 + lh;
      v8h alo = *(const v8h*)&As[mrow * LDP + hs * 8];
      v8h ahi = *(const v8h*)&As[mrow * LDP + 16 + hs * 8];
      afr[t] = __builtin_shufflevector(alo, ahi, 0,1,2,3,4,5,6,7,8,9,10,11,12,13,14,15);
      int nrow = wn * 32 + t * 16 + lh;
      v8h blo = *(const v8h*)&Bs[nrow * LDP + hs * 8];
      v8h bhi = *(const v8h*)&Bs[nrow * LDP + 16 + hs * 8];
      bfr[t] = __builtin_shufflevector(blo, bhi, 0,1,2,3,4,5,6,7,8,9,10,11,12,13,14,15);
    }
    for (int tm = 0; tm < 2; ++tm)
      for (int tn = 0; tn < 2; ++tn)
        acc[tm][tn] = __builtin_amdgcn_wmma_f32_16x16x32_f16(
            false, afr[tm], false, bfr[tn], (short)0, acc[tm][tn], false, false);
    __syncthreads();
  }

  // ---- epilogue ----  D layout: VGPR r -> M = r + 8*hs ; lane -> N = lh
  for (int tm = 0; tm < 2; ++tm) {
    int mbase = m0 + wm * 32 + tm * 16 + 8 * hs;
    if (EP == 1 && rowmax) {                       // per-row max over valid cols
      for (int r = 0; r < 8; ++r) {
        int m = mbase + r;
        if (m < M) {
          float vm = 0.f; bool any = false;
          for (int tn = 0; tn < 2; ++tn) {
            int n = n0 + wn * 32 + tn * 16 + lh;
            if (n < Nvalid) { float v = acc[tm][tn][r]; vm = any ? fmaxf(vm, v) : v; any = true; }
          }
          if (any) atomicMax(rowmax + (long long)z * sRowz + m, ordf(vm));
        }
      }
    }
    for (int tn = 0; tn < 2; ++tn) {
      int n = n0 + wn * 32 + tn * 16 + lh;
      for (int r = 0; r < 8; ++r) {
        int m = mbase + r;
        if ((ABND || BBND) && (m >= M || n >= Ncols)) continue;
        float v = acc[tm][tn][r];
        long long di = Doff + (long long)m * ldd + n;
        if (EP == 0) {
          Df16[di] = (_Float16)(v * alpha);
        } else if (EP == 1) {
          Df32[di] = v;
        } else if (EP == 2) {
          Df32[di] = v + bias[n] + resid[(long long)m * ldres + n];
        } else if (EP == 3) {
          float h = v + bias[n];
          Df16[di] = (_Float16)(0.5f * h * (1.f + erff(h * 0.70710678118654752f)));
        } else if (EP == 4) {
          float val = v + bias[n] + resid[(long long)m * ldres + n];
          int bb = m / nPerB, nnr = m - bb * nPerB;
          finalOut[((long long)bb * Ncols + n) * nPerB + nnr] = val;   // [B,C,N]
        }
      }
    }
  }
}

// ---------------------------------------------------------------------------
// Transpose [B,C,N] -> [B,N,C] with LayerNorm.  xs = raw (residual), xln = f16
// ---------------------------------------------------------------------------
__global__ __launch_bounds__(256) void ln_transpose_kernel(
    const float* __restrict__ x, const float* __restrict__ g, const float* __restrict__ bt,
    float* __restrict__ xs, _Float16* __restrict__ xln)
{
  __shared__ float tile[Cc * 33];
  const int b  = blockIdx.y;
  const int n0 = blockIdx.x * 32;
  const int tid = threadIdx.x;
  for (int i = tid; i < Cc * 32; i += 256) {
    int c = i >> 5, nl = i & 31;                       // coalesced over nl
    tile[c * 33 + nl] = x[((long long)b * Cc + c) * Nn + n0 + nl];
  }
  __syncthreads();
  const int wave = tid >> 5, lane = tid & 31;
  for (int nl = wave; nl < 32; nl += 8) {
    float vals[8], s = 0.f, s2 = 0.f;
    for (int j = 0; j < 8; ++j) {
      int c = lane + 32 * j;
      float v = tile[c * 33 + nl];
      vals[j] = v; s += v; s2 += v * v;
    }
    for (int o = 16; o; o >>= 1) { s += __shfl_xor(s, o, 32); s2 += __shfl_xor(s2, o, 32); }
    float mu  = s * (1.f / Cc);
    float var = s2 * (1.f / Cc) - mu * mu;
    float rstd = rsqrtf(var + LN_EPS);
    long long rb = ((long long)b * Nn + n0 + nl) * Cc;
    for (int j = 0; j < 8; ++j) {
      int c = lane + 32 * j;
      xs[rb + c]  = vals[j];
      xln[rb + c] = (_Float16)((vals[j] - mu) * rstd * g[c] + bt[c]);
    }
  }
}

// LN over contiguous rows [rows,256] f32 -> f16
__global__ __launch_bounds__(256) void ln_rows_kernel(
    const float* __restrict__ in, const float* __restrict__ g, const float* __restrict__ bt,
    _Float16* __restrict__ out, int rows)
{
  int wave = threadIdx.x >> 5, lane = threadIdx.x & 31;
  int row = blockIdx.x * 8 + wave;
  if (row >= rows) return;
  const float* p = in + (long long)row * Cc;
  float vals[8], s = 0.f, s2 = 0.f;
  for (int j = 0; j < 8; ++j) { float v = p[lane + 32 * j]; vals[j] = v; s += v; s2 += v * v; }
  for (int o = 16; o; o >>= 1) { s += __shfl_xor(s, o, 32); s2 += __shfl_xor(s2, o, 32); }
  float mu = s * (1.f / Cc), var = s2 * (1.f / Cc) - mu * mu;
  float rstd = rsqrtf(var + LN_EPS);
  for (int j = 0; j < 8; ++j) {
    int c = lane + 32 * j;
    out[(long long)row * Cc + c] = (_Float16)((vals[j] - mu) * rstd * g[c] + bt[c]);
  }
}

__global__ void cvt_f16_kernel(const float* __restrict__ s, _Float16* __restrict__ d, int n) {
  int i = blockIdx.x * blockDim.x + threadIdx.x;
  if (i < n) d[i] = (_Float16)s[i];
}

// projT16[d][m] = proj[m][d], zero-padded m in [266,272)
__global__ void projT_kernel(const float* __restrict__ proj, _Float16* __restrict__ pt) {
  int i = blockIdx.x * blockDim.x + threadIdx.x;
  if (i >= Dd * Mp) return;
  int d = i / Mp, m = i % Mp;
  pt[i] = (m < Mv) ? (_Float16)proj[m * Dd + d] : (_Float16)0.f;
}

// diag[row][h] = 0.5 * sum_d q[row, h*64+d]^2   (q already scaled by normalizer)
__global__ void diag_kernel(const _Float16* __restrict__ t, float* __restrict__ dg, int rows) {
  int i = blockIdx.x * blockDim.x + threadIdx.x;
  if (i >= rows * Hh) return;
  int row = i >> 2, h = i & 3;
  const _Float16* p = t + (long long)row * Cc + h * Dd;
  float s = 0.f;
  for (int d = 0; d < Dd; ++d) { float v = (float)p[d]; s += v * v; }
  dg[i] = 0.5f * s;
}

__global__ void zero_u32_kernel(unsigned* __restrict__ p, int n) {
  int i = blockIdx.x * blockDim.x + threadIdx.x;
  if (i < n) p[i] = 0u;
}

// qp/kp = ratio * (exp(dd - diag - mx) + eps), zero on padded cols
template<bool ISQ>
__global__ void exp_kernel(const float* __restrict__ dd, const unsigned* __restrict__ rowmax,
                           const float* __restrict__ mxk, const float* __restrict__ diag,
                           _Float16* __restrict__ outp)
{
  long long e = (long long)blockIdx.x * blockDim.x + threadIdx.x;
  if (e >= (long long)BH * Nn * Mp) return;
  int col = (int)(e % Mp);
  long long r = e / Mp;                 // = bh*Nn + n
  float v = 0.f;
  if (col < Mv) {
    int bh = (int)(r / Nn), n = (int)(r % Nn);
    int b = bh / Hh, h = bh % Hh;
    float mx = ISQ ? unordf(rowmax[r]) : mxk[bh];
    float dg = diag[((long long)b * Nn + n) * Hh + h];
    v = RATIO * (expf(dd[e] - dg - mx) + KERN_EPS);
  }
  outp[e] = (_Float16)v;
}

// per-(b,h) global max over rowmax[bh*Nn .. +Nn)
__global__ __launch_bounds__(256) void reduce_mxk_kernel(
    const unsigned* __restrict__ rowmax, float* __restrict__ mxk)
{
  __shared__ unsigned sm[256];
  int bh = blockIdx.x;
  unsigned m = 0u;
  for (int i = threadIdx.x; i < Nn; i += 256)
    m = umaxu(m, rowmax[(long long)bh * Nn + i]);
  sm[threadIdx.x] = m;
  __syncthreads();
  for (int s = 128; s; s >>= 1) {
    if (threadIdx.x < s) sm[threadIdx.x] = umaxu(sm[threadIdx.x], sm[threadIdx.x + s]);
    __syncthreads();
  }
  if (threadIdx.x == 0) mxk[bh] = unordf(sm[0]);
}

// Vext[bh][n][0:64]=v head cols, [64]=1 (k_sum column), [65:80]=0
__global__ void vext_kernel(const _Float16* __restrict__ v16, _Float16* __restrict__ vext) {
  long long e = (long long)blockIdx.x * blockDim.x + threadIdx.x;
  if (e >= (long long)BH * Nn * VE) return;
  int col = (int)(e % VE);
  long long r = e / VE;
  int bh = (int)(r / Nn), n = (int)(r % Nn);
  int b = bh / Hh, h = bh % Hh;
  _Float16 val = (_Float16)0.f;
  if (col < Dd)      val = v16[((long long)b * Nn + n) * Cc + h * Dd + col];
  else if (col == Dd) val = (_Float16)1.f;
  vext[e] = val;
}

// merge heads + multiply by d_inv (column 64 of out_ext)
__global__ void norm_merge_kernel(const float* __restrict__ oext, _Float16* __restrict__ a16) {
  long long e = (long long)blockIdx.x * blockDim.x + threadIdx.x;
  if (e >= BNrows * Cc) return;
  int c = (int)(e & (Cc - 1));
  long long row = e >> 8;               // b*Nn + n
  int h = c >> 6, d = c & 63;
  int b = (int)(row / Nn), n = (int)(row % Nn);
  long long base = (((long long)(b * Hh + h)) * Nn + n) * VE;
  float den = oext[base + Dd];
  a16[e] = (_Float16)(oext[base + d] / den);
}

// ---------------------------------------------------------------------------
// Host orchestration
// ---------------------------------------------------------------------------
extern "C" void kernel_launch(void* const* d_in, const int* in_sizes, int n_in,
                              void* d_out, int out_size, void* d_ws, size_t ws_size,
                              hipStream_t stream)
{
  (void)in_sizes; (void)n_in; (void)out_size; (void)ws_size;
  const float* x     = (const float*)d_in[0];
  const float* ln_g  = (const float*)d_in[1];
  const float* ln_b  = (const float*)d_in[2];
  const float* wq    = (const float*)d_in[3];
  const float* wk    = (const float*)d_in[4];
  const float* wv    = (const float*)d_in[5];
  const float* wo    = (const float*)d_in[6];
  const float* bo    = (const float*)d_in[7];
  const float* proj  = (const float*)d_in[8];
  const float* ffg   = (const float*)d_in[9];
  const float* ffb   = (const float*)d_in[10];
  const float* w1    = (const float*)d_in[11];
  const float* b1    = (const float*)d_in[12];
  const float* w2    = (const float*)d_in[13];
  const float* b2    = (const float*)d_in[14];

  char* base = (char*)d_ws; size_t off = 0;
  auto carve = [&](size_t bytes) -> char* {
    char* p = base + off;
    off = (off + bytes + 255) & ~(size_t)255;
    return p;
  };
  float*     xs     = (float*)    carve(BNrows * Cc * 4);
  _Float16*  xln16  = (_Float16*) carve(BNrows * Cc * 2);
  _Float16*  q16    = (_Float16*) carve(BNrows * Cc * 2);
  _Float16*  k16    = (_Float16*) carve(BNrows * Cc * 2);
  _Float16*  v16    = (_Float16*) carve(BNrows * Cc * 2);
  _Float16*  wq16   = (_Float16*) carve(Cc * Cc * 2);
  _Float16*  wk16   = (_Float16*) carve(Cc * Cc * 2);
  _Float16*  wv16   = (_Float16*) carve(Cc * Cc * 2);
  _Float16*  wo16   = (_Float16*) carve(Cc * Cc * 2);
  _Float16*  w116   = (_Float16*) carve(Cc * Cc * 2);
  _Float16*  w216   = (_Float16*) carve(Cc * Cc * 2);
  _Float16*  projT16= (_Float16*) carve(Dd * Mp * 2);
  float*     diagq  = (float*)    carve(BNrows * Hh * 4);
  float*     diagk  = (float*)    carve(BNrows * Hh * 4);
  float*     dd     = (float*)    carve((long long)BH * Nn * Mp * 4);
  unsigned*  rowmax = (unsigned*) carve((long long)BH * Nn * 4);
  float*     mxk    = (float*)    carve(BH * 4);
  _Float16*  qp16   = (_Float16*) carve((long long)BH * Nn * Mp * 2);
  _Float16*  kp16   = (_Float16*) carve((long long)BH * Nn * Mp * 2);
  _Float16*  vext16 = (_Float16*) carve((long long)BH * Nn * VE * 2);
  _Float16*  ctx16  = (_Float16*) carve((long long)BH * Mp * VE * 2);
  float*     oext   = (float*)    carve((long long)BH * Nn * VE * 4);
  _Float16*  ai16   = (_Float16*) carve(BNrows * Cc * 2);
  float*     attnf  = (float*)    carve(BNrows * Cc * 4);
  _Float16*  ln216  = (_Float16*) carve(BNrows * Cc * 2);
  _Float16*  h116   = (_Float16*) carve(BNrows * Cc * 2);

  const int CC2 = Cc * Cc;
  cvt_f16_kernel<<<(CC2 + 255) / 256, 256, 0, stream>>>(wq, wq16, CC2);
  cvt_f16_kernel<<<(CC2 + 255) / 256, 256, 0, stream>>>(wk, wk16, CC2);
  cvt_f16_kernel<<<(CC2 + 255) / 256, 256, 0, stream>>>(wv, wv16, CC2);
  cvt_f16_kernel<<<(CC2 + 255) / 256, 256, 0, stream>>>(wo, wo16, CC2);
  cvt_f16_kernel<<<(CC2 + 255) / 256, 256, 0, stream>>>(w1, w116, CC2);
  cvt_f16_kernel<<<(CC2 + 255) / 256, 256, 0, stream>>>(w2, w216, CC2);
  projT_kernel<<<(Dd * Mp + 255) / 256, 256, 0, stream>>>(proj, projT16);

  ln_transpose_kernel<<<dim3(Nn / 32, Bb), 256, 0, stream>>>(x, ln_g, ln_b, xs, xln16);

  // ---- QKV GEMMs (exact tiles: async-LDS fast path on both A, plain B) ----
  dim3 gFull(Cc / TBN, (int)(BNrows / TBM), 1);   // (4, 512)
  wmma_gemm<0,false,false,false><<<gFull,128,0,stream>>>(xln16, wq16, (int)BNrows, Cc, Cc, Cc, Cc, Cc,
      0,0,0,0,1, NORMALIZER, nullptr, q16, nullptr, nullptr,0, nullptr,0,0, nullptr,0);
  wmma_gemm<0,false,false,false><<<gFull,128,0,stream>>>(xln16, wk16, (int)BNrows, Cc, Cc, Cc, Cc, Cc,
      0,0,0,0,1, NORMALIZER, nullptr, k16, nullptr, nullptr,0, nullptr,0,0, nullptr,0);
  wmma_gemm<0,false,false,false><<<gFull,128,0,stream>>>(xln16, wv16, (int)BNrows, Cc, Cc, Cc, Cc, Cc,
      0,0,0,0,1, 1.0f, nullptr, v16, nullptr, nullptr,0, nullptr,0,0, nullptr,0);

  diag_kernel<<<(int)((BNrows * Hh + 255) / 256), 256, 0, stream>>>(q16, diagq, (int)BNrows);
  diag_kernel<<<(int)((BNrows * Hh + 255) / 256), 256, 0, stream>>>(k16, diagk, (int)BNrows);

  // ---- q feature map: dd = q_h @ projT (async A, bounded B + row-max) ----
  const int rmN = BH * Nn;
  dim3 gDD((Mp + TBN - 1) / TBN, Nn / TBM, BH);   // (5, 64, 32)
  zero_u32_kernel<<<(rmN + 255) / 256, 256, 0, stream>>>(rowmax, rmN);
  wmma_gemm<1,false,false,true><<<gDD,128,0,stream>>>(q16, projT16, Nn, Mp, Dd, Cc, Mp, Mp,
      (long long)Nn * Cc, Dd, 0, (long long)Nn * Mp, Hh, 1.0f,
      dd, nullptr, nullptr, nullptr,0, rowmax, Nn, Mv, nullptr,0);
  {
    long long tot = (long long)BH * Nn * Mp;
    exp_kernel<true><<<(int)((tot + 255) / 256), 256, 0, stream>>>(dd, rowmax, nullptr, diagq, qp16);
  }

  // ---- k feature map: global (per b,h) max ----
  zero_u32_kernel<<<(rmN + 255) / 256, 256, 0, stream>>>(rowmax, rmN);
  wmma_gemm<1,false,false,true><<<gDD,128,0,stream>>>(k16, projT16, Nn, Mp, Dd, Cc, Mp, Mp,
      (long long)Nn * Cc, Dd, 0, (long long)Nn * Mp, Hh, 1.0f,
      dd, nullptr, nullptr, nullptr,0, rowmax, Nn, Mv, nullptr,0);
  reduce_mxk_kernel<<<BH, 256, 0, stream>>>(rowmax, mxk);
  {
    long long tot = (long long)BH * Nn * Mp;
    exp_kernel<false><<<(int)((tot + 255) / 256), 256, 0, stream>>>(dd, nullptr, mxk, diagk, kp16);
  }

  // ---- linear attention ----
  {
    long long tot = (long long)BH * Nn * VE;
    vext_kernel<<<(int)((tot + 255) / 256), 256, 0, stream>>>(v16, vext16);
  }
  // context_ext[m, 0:80] = kp^T @ [v | 1 | 0]   (K = Nn = 4096)
  dim3 gCtx((VE + TBN - 1) / TBN, (Mp + TBM - 1) / TBM, BH);   // (2, 5, 32)
  wmma_gemm<0,true,true,true><<<gCtx,128,0,stream>>>(kp16, vext16, Mp, VE, Nn, Mp, VE, VE,
      (long long)Nn * Mp, 0, (long long)Nn * VE, (long long)Mp * VE, 1, 1.0f,
      nullptr, ctx16, nullptr, nullptr,0, nullptr,0,0, nullptr,0);
  // out_ext = qp @ context_ext  (K = Mp = 272; col 64 carries qp@k_sum)
  dim3 gOut((VE + TBN - 1) / TBN, Nn / TBM, BH);               // (2, 64, 32)
  wmma_gemm<1,false,true,true><<<gOut,128,0,stream>>>(qp16, ctx16, Nn, VE, Mp, Mp, VE, VE,
      (long long)Nn * Mp, 0, (long long)Mp * VE, (long long)Nn * VE, 1, 1.0f,
      oext, nullptr, nullptr, nullptr,0, nullptr,0,0, nullptr,0);
  {
    long long tot = BNrows * Cc;
    norm_merge_kernel<<<(int)((tot + 255) / 256), 256, 0, stream>>>(oext, ai16);
  }

  // ---- output projection + residual ----
  wmma_gemm<2,false,false,false><<<gFull,128,0,stream>>>(ai16, wo16, (int)BNrows, Cc, Cc, Cc, Cc, Cc,
      0,0,0,0,1, 1.0f, attnf, nullptr, bo, xs, Cc, nullptr,0,0, nullptr,0);

  // ---- feed-forward ----
  ln_rows_kernel<<<(int)(BNrows / 8), 256, 0, stream>>>(attnf, ffg, ffb, ln216, (int)BNrows);
  wmma_gemm<3,false,false,false><<<gFull,128,0,stream>>>(ln216, w116, (int)BNrows, Cc, Cc, Cc, Cc, Cc,
      0,0,0,0,1, 1.0f, nullptr, h116, b1, nullptr,0, nullptr,0,0, nullptr,0);
  wmma_gemm<4,false,false,false><<<gFull,128,0,stream>>>(h116, w216, (int)BNrows, Cc, Cc, Cc, Cc, Cc,
      0,0,0,0,1, 1.0f, nullptr, nullptr, b2, attnf, Cc, nullptr,0,0, (float*)d_out, Nn);
}
